// NeuralRenderer_35141422416020
// MI455X (gfx1250) — matrix operand order repaired and checked by reading, not compiled, not run on hardware
//
#include <hip/hip_runtime.h>
#include <math.h>

typedef _Float16 h16;
typedef __attribute__((ext_vector_type(16))) _Float16 v16h;
typedef __attribute__((ext_vector_type(8)))  _Float16 v8h;
typedef __attribute__((ext_vector_type(8)))  float    v8f;
typedef __attribute__((ext_vector_type(4)))  int      v4i;

#define GAS __attribute__((address_space(1)))
#define LAS __attribute__((address_space(3)))

// ---------------------------------------------------------------------------
// A-fragment K index map for 16-bit 16x32 A matrix (contiguous order):
//   element i: lanes 0-15 hold K=0..15, lanes 16-31 hold K=16..31.
__device__ __forceinline__ int kmapA(int lane, int i) {
  return (((lane >> 4) & 1) << 4) + i;
}

// Stage one 16-byte chunk from global to LDS. Prefers the CDNA5 async
// global->LDS DMA path (GLOBAL_LOAD_ASYNC_TO_LDS_B128, ASYNCcnt) when the
// builtin exists on this toolchain; falls back to a VGPR round-trip.
// Probe result (round 2): builtin exists with 4 args, param0 = int4 AS1*.
__device__ __forceinline__ void stage16(h16* __restrict__ dst,
                                        const h16* __restrict__ src, bool inb) {
#if defined(__gfx1250__) && __has_builtin(__builtin_amdgcn_global_load_async_to_lds_b128)
  if (inb) {
    __builtin_amdgcn_global_load_async_to_lds_b128(
        (GAS v4i*)(void*)(h16*)src, (LAS v4i*)(void*)dst, 0, 0);
  } else {
    v8h z;
#pragma unroll
    for (int q = 0; q < 8; ++q) z[q] = (h16)0;
    *(v8h*)dst = z;
  }
#else
  v8h val;
  if (inb) {
    val = *(const v8h*)src;
  } else {
#pragma unroll
    for (int q = 0; q < 8; ++q) val[q] = (h16)0;
  }
  *(v8h*)dst = val;
#endif
}

__device__ __forceinline__ void wait_async_stage() {
#if defined(__gfx1250__) && __has_builtin(__builtin_amdgcn_global_load_async_to_lds_b128)
#if __has_builtin(__builtin_amdgcn_s_wait_asynccnt)
  __builtin_amdgcn_s_wait_asynccnt(0);
#else
  asm volatile("s_wait_asynccnt 0" ::: "memory");
#endif
#endif
}

// h = fg + (1-mask)*bg  (16,256,32,32)
__global__ __launch_bounds__(256) void compose_k(const float* __restrict__ fg,
                                                 const float* __restrict__ mask,
                                                 const float* __restrict__ bg,
                                                 float* __restrict__ out) {
  int idx = blockIdx.x * blockDim.x + threadIdx.x;
  const int total = 16 * 256 * 32 * 32;
  if (idx >= total) return;
  int hw = idx & 1023;
  int b = idx >> 18;  // /(256*1024)
  float m = mask[b * 1024 + hw];
  out[idx] = fg[idx] + (1.0f - m) * bg[idx];
}

// style[b,ci] = z[b]·mw[ci] / sqrt(512) + mb[ci]
__global__ __launch_bounds__(256) void style_k(const float* __restrict__ z,
                                               const float* __restrict__ mw,
                                               const float* __restrict__ mb,
                                               float* __restrict__ style, int Ci) {
  int idx = blockIdx.x * blockDim.x + threadIdx.x;
  if (idx >= 16 * Ci) return;
  int b = idx / Ci, ci = idx % Ci;
  float acc = 0.f;
  for (int j = 0; j < 512; ++j) acc += z[b * 512 + j] * mw[ci * 512 + j];
  style[idx] = acc * 0.04419417382415922f + mb[ci];
}

// demod[b,co] = rsqrt( invs^2 * sum_ci style^2 * sum_tap w^2  + 1e-8 )
__global__ __launch_bounds__(256) void demod_k(const float* __restrict__ w,
                                               const float* __restrict__ style,
                                               float* __restrict__ demod,
                                               int Co, int Ci, float invs) {
  int idx = blockIdx.x * blockDim.x + threadIdx.x;
  if (idx >= 16 * Co) return;
  int b = idx / Co, co = idx % Co;
  float acc = 0.f;
  for (int ci = 0; ci < Ci; ++ci) {
    float s = style[b * Ci + ci];
    float wsum = 0.f;
    const float* wp = w + (size_t)(co * Ci + ci) * 9;
    for (int t = 0; t < 9; ++t) { float wv = wp[t]; wsum += wv * wv; }
    acc += wsum * s * s;
  }
  demod[idx] = rsqrtf(acc * invs * invs + 1e-8f);
}

// Re-pack weights (f32 OIHW) -> f16 WMMA A-fragment order:
//   aw[(((coTile*9+tap)*CiCh + cc)*32 + lane)*16 + i] = ws[co][ci][tap]
__global__ __launch_bounds__(256) void prep_w(const float* __restrict__ w,
                                              h16* __restrict__ aw,
                                              int Co, int Ci, float invs) {
  int idx = blockIdx.x * blockDim.x + threadIdx.x;
  int total = Co * Ci * 9;
  if (idx >= total) return;
  int i = idx & 15;
  int lane = (idx >> 4) & 31;
  int rest = idx >> 9;
  int CiCh = Ci >> 5;
  int cc = rest % CiCh;
  int rest2 = rest / CiCh;
  int tap = rest2 % 9;
  int coTile = rest2 / 9;
  int co = coTile * 16 + (lane & 15);
  int ci = cc * 32 + kmapA(lane, i);
  aw[idx] = (h16)(w[(size_t)(co * Ci + ci) * 9 + tap] * invs);
}

// xmod[((b*S+y)*S+x)*Ci + ci] = act_nchw * style  (f16, channel-contiguous)
__global__ __launch_bounds__(256) void modulate_k(const float* __restrict__ act,
                                                  const float* __restrict__ style,
                                                  h16* __restrict__ xm, int Ci, int S) {
  long idx = (long)blockIdx.x * blockDim.x + threadIdx.x;
  long total = (long)16 * Ci * S * S;
  if (idx >= total) return;
  int ci = (int)(idx % Ci);
  long rest = idx / Ci;
  int x = (int)(rest % S); rest /= S;
  int y = (int)(rest % S);
  int b = (int)(rest / S);
  float v = act[(((long)b * Ci + ci) * S + y) * S + x] * style[b * Ci + ci];
  xm[idx] = (h16)v;
}

// ---------------------------------------------------------------------------
// 3x3 pad-1 modulated conv, implicit GEMM via v_wmma_f32_16x16x32_f16.
// Block = one (b, y0, 32-pixel x-tile); wave = one 16-channel Co tile computing
// TWO 16-pixel N tiles so each A (weight) fragment feeds two WMMAs.
__global__ void conv3_wmma(const h16* __restrict__ xm, const h16* __restrict__ aw,
                           const float* __restrict__ demod, const float* __restrict__ bias,
                           float* __restrict__ out, int Ci, int Co, int S) {
  extern __shared__ h16 smem[];  // [3 rows][34 cols][Ci]
  const int COLS = 34;
  int tilesX = S >> 5;
  int tx = blockIdx.x % tilesX;
  int rest = blockIdx.x / tilesX;
  int y0 = rest % S;
  int b = rest / S;
  int x0 = tx * 32;

  // Cooperative LDS stage (async DMA when available; zero-fill halo OOB)
  int cvec = Ci >> 3;
  int nvec = 3 * COLS * cvec;
  for (int v = threadIdx.x; v < nvec; v += blockDim.x) {
    int ciV = v % cvec;
    int p = v / cvec;
    int c = p % COLS;
    int r = p / COLS;
    int y = y0 - 1 + r, x = x0 - 1 + c;
    bool inb = (y >= 0 && y < S && x >= 0 && x < S);
    const h16* src = xm + (((long)(b * S + y) * S + x) * Ci) + ciV * 8;
    stage16(smem + (long)v * 8, src, inb);
  }
  wait_async_stage();
  __syncthreads();

  int wave = threadIdx.x >> 5;     // Co tile
  int lane = threadIdx.x & 31;
  int n = lane & 15;               // output pixel in sub-tile
  int hi = (lane >> 4) & 1;
  int co0 = wave * 16;
  int CiCh = Ci >> 5;
  const h16* awTile = aw + (long)wave * 9 * CiCh * 512;

  v8f acc0 = {};
  v8f acc1 = {};
  for (int cc = 0; cc < CiCh; ++cc) {
    if (cc + 1 < CiCh)
      __builtin_prefetch(awTile + ((long)(cc + 1)) * 512, 0, 1);  // global_prefetch_b8
#pragma unroll
    for (int tap = 0; tap < 9; ++tap) {
      int kh = tap / 3, kw = tap % 3;
      v16h a = *(const v16h*)(awTile + (((long)tap * CiCh + cc) * 512) + lane * 16);
      const h16* bbase = smem + ((long)kh * COLS * Ci) + cc * 32 + hi * 16;
      v16h bf0 = *(const v16h*)(bbase + (long)(n + kw) * Ci);
      v16h bf1 = *(const v16h*)(bbase + (long)(n + 16 + kw) * Ci);
      acc0 = __builtin_amdgcn_wmma_f32_16x16x32_f16(false, a, false, bf0, (short)0, acc0,
                                                    false, false);
      acc1 = __builtin_amdgcn_wmma_f32_16x16x32_f16(false, a, false, bf1, (short)0, acc1,
                                                    false, false);
    }
  }

  const float SQ2 = 1.4142135623730951f;
#pragma unroll
  for (int r = 0; r < 8; ++r) {
    int co = co0 + r + hi * 8;
    float dm = demod[b * Co + co];
    float bs = bias[co];
    float* orow = out + (((long)b * Co + co) * S + y0) * S;
    float v0 = acc0[r] * dm + bs;
    float v1 = acc1[r] * dm + bs;
    orow[x0 + n]      = (v0 > 0.f ? v0 : 0.2f * v0) * SQ2;
    orow[x0 + 16 + n] = (v1 > 0.f ? v1 : 0.2f * v1) * SQ2;
  }
}

// ---------------------------------------------------------------------------
// Transposed 3x3 stride-2 conv (StyleGAN up), gathered form:
//   out[oy,ox] = sum_{ky,kx: (oy-ky),(ox-kx) even} ws[ky,kx]*x[(oy-ky)/2,(ox-kx)/2]
// Output grid (2S+1)^2; demod applied here (bias/act come after blur).
__global__ void tconv3_wmma(const h16* __restrict__ xm, const h16* __restrict__ aw,
                            const float* __restrict__ demod,
                            float* __restrict__ out, int Ci, int Co, int S) {
  extern __shared__ h16 smem[];  // [2 rows][9 cols][Ci]
  const int COLS = 9;
  int So = 2 * S + 1;
  int tilesX = (So + 15) >> 4;
  int tx = blockIdx.x % tilesX;
  int rest = blockIdx.x / tilesX;
  int oy = rest % So;
  int b = rest / So;
  int x0 = tx * 16;
  int ybase = (oy >> 1) - 1;
  int xibase = (x0 >> 1) - 1;

  int cvec = Ci >> 3;
  int nvec = 2 * COLS * cvec;
  for (int v = threadIdx.x; v < nvec; v += blockDim.x) {
    int ciV = v % cvec;
    int p = v / cvec;
    int c = p % COLS;
    int r = p / COLS;
    int y = ybase + r, x = xibase + c;
    bool inb = (y >= 0 && y < S && x >= 0 && x < S);
    const h16* src = xm + (((long)(b * S + y) * S + x) * Ci) + ciV * 8;
    stage16(smem + (long)v * 8, src, inb);
  }
  wait_async_stage();
  __syncthreads();

  int wave = threadIdx.x >> 5;
  int lane = threadIdx.x & 31;
  int n = lane & 15;
  int hi = (lane >> 4) & 1;
  int co0 = wave * 16;
  int ox = x0 + n;
  int CiCh = Ci >> 5;
  const h16* awTile = aw + (long)wave * 9 * CiCh * 512;

  v8f acc = {};
  for (int cc = 0; cc < CiCh; ++cc) {
#pragma unroll
    for (int tap = 0; tap < 9; ++tap) {
      int kh = tap / 3, kw = tap % 3;
      int ty = oy - kh;
      if (ty & 1) continue;            // uniform across wave
      int iy = ty >> 1;
      if (iy < 0 || iy >= S) continue; // uniform across wave
      int r = iy - ybase;
      // per-lane column validity (parity + range); masked lanes feed zeros
      int txx = ox - kw;
      bool lv = ((txx & 1) == 0);
      int ix = txx >> 1;
      lv = lv && (ix >= 0) && (ix < S);
      int c = lv ? (ix - xibase) : 0;
      v16h a = *(const v16h*)(awTile + (((long)tap * CiCh + cc) * 512) + lane * 16);
      v16h bf = *(const v16h*)(smem + ((long)(r * COLS + c) * Ci) + cc * 32 + hi * 16);
      if (!lv) {
#pragma unroll
        for (int q = 0; q < 16; ++q) bf[q] = (h16)0;
      }
      acc = __builtin_amdgcn_wmma_f32_16x16x32_f16(false, a, false, bf, (short)0, acc,
                                                   false, false);
    }
  }

#pragma unroll
  for (int r = 0; r < 8; ++r) {
    int co = co0 + r + hi * 8;
    if (ox < So)
      out[(((long)b * Co + co) * So + oy) * So + ox] = acc[r] * demod[b * Co + co];
  }
}

// Depthwise 4x4 binomial blur (pad 1): (2S+1)^2 -> (2S)^2, fused bias+lrelu*sqrt2
__global__ __launch_bounds__(256) void blur_k(const float* __restrict__ in,
                                              const float* __restrict__ bias,
                                              float* __restrict__ out, int C, int Sout) {
  int Sin = Sout + 1;
  long idx = (long)blockIdx.x * blockDim.x + threadIdx.x;
  long total = (long)16 * C * Sout * Sout;
  if (idx >= total) return;
  int x = (int)(idx % Sout);
  long rest = idx / Sout;
  int y = (int)(rest % Sout); rest /= Sout;
  int c = (int)(rest % C);
  int b = (int)(rest / C);
  const float wk[4] = {1.f, 3.f, 3.f, 1.f};
  float acc = 0.f;
  const float* base = in + ((long)b * C + c) * Sin * Sin;
#pragma unroll
  for (int i = 0; i < 4; ++i) {
    int yy = y + i - 1;
    if (yy < 0 || yy >= Sin) continue;
#pragma unroll
    for (int j = 0; j < 4; ++j) {
      int xx = x + j - 1;
      if (xx < 0 || xx >= Sin) continue;
      acc += wk[i] * wk[j] * base[(long)yy * Sin + xx];
    }
  }
  float v = acc * 0.0625f + bias[c];
  out[idx] = (v > 0.f ? v : 0.2f * v) * 1.4142135623730951f;
}

// Final modulated 1x1 conv 32->4 (no demod), + out_bias, keep channels 0..2
__global__ __launch_bounds__(256) void final_k(const float* __restrict__ act,
                                               const float* __restrict__ style,
                                               const float* __restrict__ fw,
                                               const float* __restrict__ ob,
                                               float* __restrict__ out) {
  int idx = blockIdx.x * blockDim.x + threadIdx.x;
  const int total = 16 * 3 * 128 * 128;
  if (idx >= total) return;
  int x = idx & 127;
  int y = (idx >> 7) & 127;
  int rest = idx >> 14;
  int c = rest % 3;
  int b = rest / 3;
  float acc = 0.f;
  for (int ci = 0; ci < 32; ++ci)
    acc += fw[c * 32 + ci] * style[b * 32 + ci] *
           act[(((long)b * 32 + ci) * 128 + y) * 128 + x];
  out[idx] = acc * 0.17677669529663687f + ob[c];  // 1/sqrt(32)
}

// ---------------------------------------------------------------------------
extern "C" void kernel_launch(void* const* d_in, const int* in_sizes, int n_in,
                              void* d_out, int out_size, void* d_ws, size_t ws_size,
                              hipStream_t stream) {
  const float* fg   = (const float*)d_in[0];
  const float* mask = (const float*)d_in[1];
  const float* bg   = (const float*)d_in[2];
  const float* z    = (const float*)d_in[3];
  const float *lw[5], *lmw[5], *lmb[5], *lb[5];
  for (int l = 0; l < 5; ++l) {
    lw[l]  = (const float*)d_in[4 + 4 * l];
    lmw[l] = (const float*)d_in[5 + 4 * l];
    lmb[l] = (const float*)d_in[6 + 4 * l];
    lb[l]  = (const float*)d_in[7 + 4 * l];
  }
  const float* fin_w    = (const float*)d_in[24];
  const float* fin_mw   = (const float*)d_in[25];
  const float* fin_mb   = (const float*)d_in[26];
  const float* out_bias = (const float*)d_in[27];

  const size_t ACT_BYTES = 68161536;  // 17,040,384 f32 (16*64*129*129)
  const size_t XM_BYTES  = 33554432;  // 16,777,216 f16 (16*64*128*128)
  const size_t AW_BYTES  = 1048576;
  const size_t NEED = 2 * ACT_BYTES + XM_BYTES + AW_BYTES + 32768;
  if (ws_size < NEED) return;

  char* ws = (char*)d_ws;
  float* bufA   = (float*)ws;
  float* bufB   = (float*)(ws + ACT_BYTES);
  h16*   xm     = (h16*)(ws + 2 * ACT_BYTES);
  h16*   awb    = (h16*)(ws + 2 * ACT_BYTES + XM_BYTES);
  float* styleB = (float*)(ws + 2 * ACT_BYTES + XM_BYTES + AW_BYTES);
  float* demodB = styleB + 4096;

  auto cdiv = [](long a, long b) { return (unsigned)((a + b - 1) / b); };

  { long n = 16L * 256 * 32 * 32;
    compose_k<<<cdiv(n, 256), 256, 0, stream>>>(fg, mask, bg, bufA); }

  const int  Cis[5] = {256, 128, 128, 64, 64};
  const int  Cos[5] = {128, 128, 64, 64, 32};
  const bool ups[5] = {false, true, false, true, false};
  float* act = bufA;
  float* other = bufB;
  int S = 32;

  for (int l = 0; l < 5; ++l) {
    int Ci = Cis[l], Co = Cos[l];
    float invs = 1.0f / sqrtf((float)(Ci * 9));
    style_k<<<cdiv(16L * Ci, 256), 256, 0, stream>>>(z, lmw[l], lmb[l], styleB, Ci);
    demod_k<<<cdiv(16L * Co, 256), 256, 0, stream>>>(lw[l], styleB, demodB, Co, Ci, invs);
    { long n = (long)Co * Ci * 9;
      prep_w<<<cdiv(n, 256), 256, 0, stream>>>(lw[l], awb, Co, Ci, invs); }
    { long n = 16L * Ci * S * S;
      modulate_k<<<cdiv(n, 256), 256, 0, stream>>>(act, styleB, xm, Ci, S); }

    if (!ups[l]) {
      dim3 grid(16u * S * (S / 32));
      dim3 blk(32u * (Co / 16));
      size_t shm = (size_t)3 * 34 * Ci * sizeof(h16);
      conv3_wmma<<<grid, blk, shm, stream>>>(xm, awb, demodB, lb[l], other, Ci, Co, S);
      float* t = act; act = other; other = t;
    } else {
      int So = 2 * S + 1;
      int tilesX = (So + 15) / 16;
      dim3 grid((unsigned)(16 * So * tilesX));
      dim3 blk(32u * (Co / 16));
      size_t shm = (size_t)2 * 9 * Ci * sizeof(h16);
      tconv3_wmma<<<grid, blk, shm, stream>>>(xm, awb, demodB, other, Ci, Co, S);
      long n = 16L * Co * (2 * S) * (2 * S);
      blur_k<<<cdiv(n, 256), 256, 0, stream>>>(other, lb[l], act, Co, 2 * S);
      S = 2 * S;
    }
  }

  style_k<<<cdiv(16L * 32, 256), 256, 0, stream>>>(z, fin_mw, fin_mb, styleB, 32);
  { long n = 16L * 3 * 128 * 128;
    final_k<<<cdiv(n, 256), 256, 0, stream>>>(act, styleB, fin_w, out_bias,
                                              (float*)d_out); }
}
